// MultiHeadedAttention_21620865368064
// MI455X (gfx1250) — matrix operand order repaired and checked by reading, not compiled, not run on hardware
//
#include <hip/hip_runtime.h>

// ---------------------------------------------------------------------------
// MI455X (gfx1250) top-k sparse MHA.  B=2, S=2048, D=1024, H=16, DK=64, K=32.
// Pipeline:
//   0) f32->bf16 conversion pre-pass for activations + weights (bandwidth-trivial)
//   1) gemm_nt_bias<bf16 out> x3 : Q/K/V projections; tiles staged into LDS by
//      the Tensor Data Mover (tensor_load_to_lds, TENSORcnt), math via
//      v_wmma_f32_16x16x32_bf16 (fp32 accumulate)
//   2) attn_topk : WMMA scores -> LDS chunk -> wave top-32 -> softmax -> P@V
//   3) gemm_nt_bias<f32 out> : output projection -> d_out
// ---------------------------------------------------------------------------

typedef unsigned short u16;
typedef unsigned int   u32;
typedef unsigned long long u64;

typedef __attribute__((ext_vector_type(16))) __bf16 v16bf;
typedef __attribute__((ext_vector_type(8)))  float  v8f;
typedef __attribute__((ext_vector_type(4)))  u32    u32x4;
typedef __attribute__((ext_vector_type(8)))  u32    u32x8;

union FragBF {           // 32 bytes: one 16x16x32 bf16 WMMA operand per lane
  uint4 u[2];
  v16bf v;
};

static __device__ inline v8f zero8() {
  v8f z;
#pragma unroll
  for (int i = 0; i < 8; ++i) z[i] = 0.0f;
  return z;
}

static __device__ inline v8f wmma_bf16(v16bf a, v16bf b, v8f c) {
  return __builtin_amdgcn_wmma_f32_16x16x32_bf16(false, a, false, b, (short)0, c,
                                                 false, false);
}

// fp32 -> bf16 (round to nearest even)
static __device__ inline u16 f2bf(float f) {
  u32 u = __float_as_uint(f);
  u32 r = u + 0x7FFFu + ((u >> 16) & 1u);
  return (u16)(r >> 16);
}
static __device__ inline float bf2f(u32 lo16) {
  return __uint_as_float(lo16 << 16);
}

// ---------------------------------------------------------------------------
// Tensor Data Mover: DMA a 2D bf16 tile (tile_d0 contiguous elems x tile_d1
// rows, row stride = stride_d0 elems) from global into LDS at lds_off.
// D# built per CDNA5 ISA §8.3/8.4 (group0 128b, group1 256b; groups 2/3 zero).
// Issued once per wave (EXEC ignored); tracked with TENSORcnt.
// ---------------------------------------------------------------------------
static __device__ inline void tdm_load_2d_bf16(u32 lds_off, const u16* gsrc,
                                               u32 tile_d0, u32 tile_d1,
                                               u32 tensor_d0, u32 tensor_d1,
                                               u32 stride_d0) {
  const u64 ga = (u64)(uintptr_t)gsrc;
  u32x4 g0;
  g0[0] = 1u;                                   // count=1, user-mode, no gather
  g0[1] = lds_off;                              // LDS byte address
  g0[2] = (u32)ga;                              // global_addr[31:0]
  g0[3] = (u32)((ga >> 32) & 0x01FFFFFFu) | (2u << 30);  // addr[56:32], type=2
  u32x8 g1;
  g1[0] = (1u << 16);                           // data_size=1 (2 bytes)
  g1[1] = (tensor_d0 & 0xFFFFu) << 16;          // tensor_dim0[15:0]
  g1[2] = (tensor_d0 >> 16) | ((tensor_d1 & 0xFFFFu) << 16);
  g1[3] = (tensor_d1 >> 16) | (tile_d0 << 16);  // tile_dim0
  g1[4] = tile_d1;                              // tile_dim1 (tile_dim2=0)
  g1[5] = stride_d0;                            // tensor_dim0_stride[31:0]
  g1[6] = 0;
  g1[7] = 0;
  const u32x4 gz = {0u, 0u, 0u, 0u};            // groups 2/3: dims >2 unused
  asm volatile("tensor_load_to_lds %0, %1, %2, %3"
               :
               : "s"(g0), "s"(g1), "s"(gz), "s"(gz)
               : "memory");
}

// ---------------------------------------------------------------------------
// fp32 -> bf16 elementwise conversion (8 elems/thread)
// ---------------------------------------------------------------------------
__global__ __launch_bounds__(256) void f32_to_bf16_kernel(
    const float* __restrict__ src, u16* __restrict__ dst, int n8) {
  const int i = blockIdx.x * blockDim.x + threadIdx.x;
  if (i >= n8) return;
  const float4* s = (const float4*)src + (size_t)i * 2;
  const float4 f0 = s[0], f1 = s[1];
  alignas(16) u16 tmp[8] = {f2bf(f0.x), f2bf(f0.y), f2bf(f0.z), f2bf(f0.w),
                            f2bf(f1.x), f2bf(f1.y), f2bf(f1.z), f2bf(f1.w)};
  ((uint4*)dst)[i] = *(uint4*)tmp;
}

// ---------------------------------------------------------------------------
// GEMM: C[M,N] = A[M,Kd] * W[N,Kd]^T + bias[N]   (A, W bf16; accumulate f32)
// Tile 128x128, 256 threads (8 waves), wave sub-tile 32x64 (2x4 frags).
// Double-buffered LDS tiles staged by the TDM (wave 0 issues, TENSORcnt).
// ---------------------------------------------------------------------------
template <bool OUT_BF16>
__global__ __launch_bounds__(256) void gemm_nt_bias_kernel(
    const u16* __restrict__ A, const u16* __restrict__ W,
    const float* __restrict__ bias, void* __restrict__ Cout,
    int M, int N, int Kd) {
  __shared__ u16 As[2][128 * 32];   // 8 KB per buffer
  __shared__ u16 Bs[2][128 * 32];

  const int tiles_n = N >> 7;
  const int tm = (int)blockIdx.x / tiles_n;
  const int tn = (int)blockIdx.x % tiles_n;
  const int t = threadIdx.x;
  const int w = t >> 5, lane = t & 31;
  const int wm = w >> 1, wn = w & 1;      // 4x2 wave grid
  const int lm = lane & 15, hs = lane >> 4;

  v8f acc[2][4];
#pragma unroll
  for (int mi = 0; mi < 2; ++mi)
#pragma unroll
    for (int ni = 0; ni < 4; ++ni) acc[mi][ni] = zero8();

  auto stage_tdm = [&](int buf, int kk) {
    tdm_load_2d_bf16((u32)(uintptr_t)&As[buf][0],
                     A + (size_t)(tm * 128) * Kd + kk,
                     /*tile*/ 32, 128, /*tensor*/ (u32)Kd, (u32)M, (u32)Kd);
    tdm_load_2d_bf16((u32)(uintptr_t)&Bs[buf][0],
                     W + (size_t)(tn * 128) * Kd + kk,
                     /*tile*/ 32, 128, /*tensor*/ (u32)Kd, (u32)N, (u32)Kd);
  };

  auto compute = [&](int buf) {
    FragBF a[2], b[4];
#pragma unroll
    for (int mi = 0; mi < 2; ++mi) {
      const int row = wm * 32 + mi * 16 + lm;
      // A 16x32 bf16 layout: per lane two contiguous 8-half chunks
      a[mi].u[0] = *(const uint4*)&As[buf][row * 32 + hs * 8];
      a[mi].u[1] = *(const uint4*)&As[buf][row * 32 + 16 + hs * 8];
    }
#pragma unroll
    for (int ni = 0; ni < 4; ++ni) {
      const int nrow = wn * 64 + ni * 16 + lm;
      // B 32x16 bf16 layout: per lane two contiguous 8-half chunks (same k order as A)
      b[ni].u[0] = *(const uint4*)&Bs[buf][nrow * 32 + hs * 8];
      b[ni].u[1] = *(const uint4*)&Bs[buf][nrow * 32 + 16 + hs * 8];
    }
#pragma unroll
    for (int mi = 0; mi < 2; ++mi)
#pragma unroll
      for (int ni = 0; ni < 4; ++ni)
        acc[mi][ni] = wmma_bf16(a[mi].v, b[ni].v, acc[mi][ni]);
  };

  if (t < 32) {                       // wave 0 drives the DMA
    stage_tdm(0, 0);
    __builtin_amdgcn_s_wait_tensorcnt(0);
  }
  __syncthreads();
  for (int kk = 0; kk < Kd; kk += 32) {
    const int buf = (kk >> 5) & 1;
    const bool more = (kk + 32) < Kd;
    if (more && t < 32) stage_tdm(buf ^ 1, kk + 32);   // overlap DMA w/ WMMA
    compute(buf);
    if (more && t < 32) __builtin_amdgcn_s_wait_tensorcnt(0);
    __syncthreads();
  }

  // epilogue: C frag -> row = base + 8*(lane/16) + g, col = base + lane%16
#pragma unroll
  for (int mi = 0; mi < 2; ++mi) {
#pragma unroll
    for (int ni = 0; ni < 4; ++ni) {
      const int col = tn * 128 + wn * 64 + ni * 16 + lm;
      const float bv = bias[col];
#pragma unroll
      for (int g = 0; g < 8; ++g) {
        const int row = tm * 128 + wm * 32 + mi * 16 + hs * 8 + g;
        const float v = acc[mi][ni][g] + bv;
        if constexpr (OUT_BF16)
          ((u16*)Cout)[(size_t)row * N + col] = f2bf(v);
        else
          ((float*)Cout)[(size_t)row * N + col] = v;
      }
    }
  }
}

// ---------------------------------------------------------------------------
// Attention with top-32 sparsification. One workgroup per (b, h, 16-q block).
// ---------------------------------------------------------------------------
#define S_LEN 2048
#define D_MODEL 1024
#define N_HEADS 16
#define DK 64
#define QT 16
#define CHUNK 512
#define NCHUNK (S_LEN / CHUNK)  // 4
#define TOPK 32

// Wave-cooperative: extract top-32 (descending) from vals[0..n), destroying it.
static __device__ inline void wave_top32(float* vals, const int* idxArr, int n,
                                         int lane, int idx_off, float* outv,
                                         int* outi) {
  for (int sel = 0; sel < TOPK; ++sel) {
    float best = -3.0e38f;
    int bi = 0x7fffffff;
    for (int j = lane; j < n; j += 32) {
      float v = vals[j];
      if (v > best || (v == best && j < bi)) { best = v; bi = j; }
    }
#pragma unroll
    for (int off = 16; off; off >>= 1) {
      float ov = __shfl_xor(best, off, 32);
      int oi = __shfl_xor(bi, off, 32);
      if (ov > best || (ov == best && oi < bi)) { best = ov; bi = oi; }
    }
    if (lane == 0) {
      outv[sel] = best;
      outi[sel] = idxArr ? idxArr[bi] : (idx_off + bi);
    }
    vals[bi] = -3.0e38f;  // all lanes agree on bi; same-address same-value store
  }
}

__global__ __launch_bounds__(256) void attn_topk_kernel(
    const u16* __restrict__ Qp, const u16* __restrict__ Kp,
    const u16* __restrict__ Vp, const int* __restrict__ mask,
    u16* __restrict__ Ctx) {
  __shared__ float sc[16][CHUNK];              // 32 KB score slab
  __shared__ u16 qs[16 * DK];                  // 2 KB Q tile
  __shared__ float candv[16][NCHUNK * TOPK];   // 8 KB
  __shared__ int candi[16][NCHUNK * TOPK];     // 8 KB
  __shared__ float prob[16][TOPK];             // 2 KB
  __shared__ int pidx[16][TOPK];               // 2 KB

  const int qtiles = S_LEN / QT;  // 128
  const int bid = blockIdx.x;
  const int qt = bid % qtiles;
  const int h = (bid / qtiles) % N_HEADS;
  const int b = bid / (qtiles * N_HEADS);

  const int t = threadIdx.x;
  const int w = t >> 5, lane = t & 31;
  const int lm = lane & 15, hs = lane >> 4;

  // stage 16x64 bf16 Q tile via the Tensor Data Mover
  if (t < 32) {
    tdm_load_2d_bf16((u32)(uintptr_t)&qs[0],
                     Qp + ((size_t)(b * S_LEN + qt * QT) * D_MODEL + h * DK),
                     /*tile*/ DK, QT, /*tensor*/ D_MODEL, QT, D_MODEL);
    __builtin_amdgcn_s_wait_tensorcnt(0);
  }
  __syncthreads();

  // per-wave Q fragments (A operand, 16x32 each, d = [0,32) and [32,64))
  FragBF fa[2];
#pragma unroll
  for (int s = 0; s < 2; ++s) {
    fa[s].u[0] = *(const uint4*)&qs[lm * DK + s * 32 + hs * 8];
    fa[s].u[1] = *(const uint4*)&qs[lm * DK + s * 32 + 16 + hs * 8];
  }

  for (int ci = 0; ci < NCHUNK; ++ci) {
    const int col0 = ci * CHUNK;
    // 32 key tiles per chunk, 4 per wave: scores via WMMA
    for (int j = 0; j < 4; ++j) {
      const int colt = col0 + (w * 4 + j) * 16;
      const int key = colt + lm;  // B operand: column = lane%16
      v8f acc = zero8();
#pragma unroll
      for (int s = 0; s < 2; ++s) {
        const u16* kp = Kp + ((size_t)(b * S_LEN + key) * D_MODEL + h * DK +
                              s * 32 + hs * 8);
        FragBF fb;
        fb.u[0] = *(const uint4*)kp;
        fb.u[1] = *(const uint4*)(kp + 16);
        acc = wmma_bf16(fa[s].v, fb.v, acc);
      }
      const int col = colt + lm;
#pragma unroll
      for (int g = 0; g < 8; ++g) {
        const int m = g + hs * 8;
        const int qrow = qt * QT + m;
        const int mk = mask[(size_t)(b * S_LEN + qrow) * S_LEN + col];
        sc[m][col - col0] = mk ? acc[g] * 0.125f : -1e9f;  // 1/sqrt(64)
      }
    }
    __syncthreads();
    // per-row chunk top-32 (rows 2w, 2w+1 per wave)
    for (int rr = 0; rr < 2; ++rr) {
      const int r = w * 2 + rr;
      wave_top32(&sc[r][0], nullptr, CHUNK, lane, col0, &candv[r][ci * TOPK],
                 &candi[r][ci * TOPK]);
    }
    __syncthreads();
  }

  // final merge of 128 candidates + softmax over the 32 survivors
  for (int rr = 0; rr < 2; ++rr) {
    const int r = w * 2 + rr;
    wave_top32(&candv[r][0], &candi[r][0], NCHUNK * TOPK, lane, 0, &prob[r][0],
               &pidx[r][0]);
    const float v = prob[r][lane];   // lane j holds the j-th largest score
    const float vmax = prob[r][0];
    const float e = __expf(v - vmax);
    float ssum = e;
#pragma unroll
    for (int off = 16; off; off >>= 1) ssum += __shfl_xor(ssum, off, 32);
    prob[r][lane] = e / ssum;
  }
  __syncthreads();

  // ctx[m, d] = sum_j prob[m][j] * V[idx[m][j], d]  (sparse gather of V)
  {
    const int m = t >> 4;            // 0..15
    const int dbase = (t & 15) * 4;  // 0..60
    float a0 = 0.f, a1 = 0.f, a2 = 0.f, a3 = 0.f;
#pragma unroll 4
    for (int j = 0; j < TOPK; ++j) {
      const float p = prob[m][j];
      const int key = pidx[m][j];
      const u16* vp =
          Vp + ((size_t)(b * S_LEN + key) * D_MODEL + h * DK + dbase);
      const uint2 raw = *(const uint2*)vp;
      a0 = fmaf(p, bf2f(raw.x & 0xFFFFu), a0);
      a1 = fmaf(p, __uint_as_float(raw.x & 0xFFFF0000u), a1);
      a2 = fmaf(p, bf2f(raw.y & 0xFFFFu), a2);
      a3 = fmaf(p, __uint_as_float(raw.y & 0xFFFF0000u), a3);
    }
    alignas(8) u16 ot[4] = {f2bf(a0), f2bf(a1), f2bf(a2), f2bf(a3)};
    *(uint2*)&Ctx[(size_t)(b * S_LEN + qt * QT + m) * D_MODEL + h * DK +
                  dbase] = *(uint2*)ot;
  }
}

// ---------------------------------------------------------------------------
// Launcher
// ---------------------------------------------------------------------------
extern "C" void kernel_launch(void* const* d_in, const int* in_sizes, int n_in,
                              void* d_out, int out_size, void* d_ws,
                              size_t ws_size, hipStream_t stream) {
  (void)in_sizes; (void)n_in; (void)out_size; (void)ws_size;
  const float* query = (const float*)d_in[0];
  const float* key   = (const float*)d_in[1];
  const float* value = (const float*)d_in[2];
  const float* Wq = (const float*)d_in[3];
  const float* bq = (const float*)d_in[4];
  const float* Wk = (const float*)d_in[5];
  const float* bk = (const float*)d_in[6];
  const float* Wv = (const float*)d_in[7];
  const float* bv = (const float*)d_in[8];
  const float* Wo = (const float*)d_in[9];
  const float* bo = (const float*)d_in[10];
  const int* mask = (const int*)d_in[11];
  // d_in[12] = topk (fixed at 32; kernels specialized for K=32)

  const int Bn = 2, S = 2048, D = 1024;
  const int M = Bn * S;  // 4096
  const size_t MB = (size_t)1 << 20;

  char* ws = (char*)d_ws;
  u16* qin = (u16*)(ws + 0 * MB);    // bf16 copies of activations (8 MB each)
  u16* kin = (u16*)(ws + 8 * MB);
  u16* vin = (u16*)(ws + 16 * MB);
  u16* wqb = (u16*)(ws + 24 * MB);   // bf16 weights (2 MB each)
  u16* wkb = (u16*)(ws + 26 * MB);
  u16* wvb = (u16*)(ws + 28 * MB);
  u16* wob = (u16*)(ws + 30 * MB);
  u16* Qp  = (u16*)(ws + 32 * MB);   // projected Q/K/V, bf16 (8 MB each)
  u16* Kp  = (u16*)(ws + 40 * MB);
  u16* Vp  = (u16*)(ws + 48 * MB);
  u16* Ctx = (u16*)(ws + 56 * MB);   // attention context, bf16 (8 MB)

  dim3 blk(256);
  const int nAct8 = (M * D) / 8;     // 524288
  const int nW8 = (D * D) / 8;       // 131072
  f32_to_bf16_kernel<<<dim3((nAct8 + 255) / 256), blk, 0, stream>>>(query, qin, nAct8);
  f32_to_bf16_kernel<<<dim3((nAct8 + 255) / 256), blk, 0, stream>>>(key,   kin, nAct8);
  f32_to_bf16_kernel<<<dim3((nAct8 + 255) / 256), blk, 0, stream>>>(value, vin, nAct8);
  f32_to_bf16_kernel<<<dim3((nW8 + 255) / 256),   blk, 0, stream>>>(Wq, wqb, nW8);
  f32_to_bf16_kernel<<<dim3((nW8 + 255) / 256),   blk, 0, stream>>>(Wk, wkb, nW8);
  f32_to_bf16_kernel<<<dim3((nW8 + 255) / 256),   blk, 0, stream>>>(Wv, wvb, nW8);
  f32_to_bf16_kernel<<<dim3((nW8 + 255) / 256),   blk, 0, stream>>>(Wo, wob, nW8);

  dim3 ggemm((unsigned)((M / 128) * (D / 128)));  // 256 workgroups
  gemm_nt_bias_kernel<true><<<ggemm, blk, 0, stream>>>(qin, wqb, bq, Qp, M, D, D);
  gemm_nt_bias_kernel<true><<<ggemm, blk, 0, stream>>>(kin, wkb, bk, Kp, M, D, D);
  gemm_nt_bias_kernel<true><<<ggemm, blk, 0, stream>>>(vin, wvb, bv, Vp, M, D, D);

  dim3 gattn((unsigned)(Bn * N_HEADS * (S / QT)));  // 4096 workgroups
  attn_topk_kernel<<<gattn, blk, 0, stream>>>(Qp, Kp, Vp, mask, Ctx);

  gemm_nt_bias_kernel<false><<<ggemm, blk, 0, stream>>>(Ctx, wob, bo, d_out, M, D, D);
}
